// ChebClassifier_41566693490715
// MI455X (gfx1250) — matrix operand order repaired and checked by reading, not compiled, not run on hardware
//
#include <hip/hip_runtime.h>
#include <hip/hip_bf16.h>

// ---------------- problem constants (from reference) ----------------
static constexpr int kN0 = 100000, kN1 = 25000, kN2 = 6250;
static constexpr int kE0 = 600000, kE1 = 150000, kE2 = 40000;
static constexpr int kK  = 6;
static constexpr int kCls = 10;
// row counts padded to 16 so WMMA tiles need no guards
static constexpr int kN1p = ((kN1 + 15) / 16) * 16;   // 25008
static constexpr int kN2p = ((kN2 + 15) / 16) * 16;   // 6256

typedef __attribute__((ext_vector_type(16))) _Float16 v16h;
typedef __attribute__((ext_vector_type(8)))  float    v8f;

// ---------------- elementwise / setup kernels ----------------
__global__ void zero_kernel(float* __restrict__ p, int n) {
  int i = blockIdx.x * blockDim.x + threadIdx.x;
  if (i < n) p[i] = 0.f;
}

__global__ void deg_kernel(const int* __restrict__ rows, float* __restrict__ deg, int E) {
  int e = blockIdx.x * blockDim.x + threadIdx.x;
  if (e < E) atomicAdd(&deg[rows[e]], 1.f);
}

__global__ void dinv_kernel(float* __restrict__ d, int n) {
  int i = blockIdx.x * blockDim.x + threadIdx.x;
  if (i < n) { float v = d[i]; d[i] = (v > 0.f) ? rsqrtf(v) : 0.f; }
}

// copy pos[n,3] into plane 0 (columns 0..2) of TxAll [n, stride]
__global__ void seed0_kernel(const float* __restrict__ pos, float* __restrict__ tx,
                             int n, int stride) {
  int t = blockIdx.x * blockDim.x + threadIdx.x;
  if (t < n * 3) tx[(t / 3) * stride + (t % 3)] = pos[t];
}

// scatter: tx[:, dstOff..dstOff+C) += w_e * tx[col, srcOff..srcOff+C)
__global__ void prop_kernel(const int* __restrict__ rows, const int* __restrict__ cols,
                            const float* __restrict__ dinv, float* __restrict__ tx,
                            int E, int C, int stride, int srcOff, int dstOff) {
  int t = blockIdx.x * blockDim.x + threadIdx.x;
  if (t >= E * C) return;
  int e = t / C, c = t - e * C;
  int r = rows[e], cl = cols[e];
  float w = -dinv[r] * dinv[cl];
  atomicAdd(&tx[r * stride + dstOff + c], w * tx[cl * stride + srcOff + c]);
}

// Tx_k = 2*prop(Tx_{k-1}) - Tx_{k-2}  (the scatter already produced prop(...))
__global__ void fixup_kernel(float* __restrict__ tx, int n, int C, int stride,
                             int dstOff, int refOff) {
  int t = blockIdx.x * blockDim.x + threadIdx.x;
  if (t >= n * C) return;
  int i = t / C, c = t - i * C;
  float* p = tx + i * stride;
  p[dstOff + c] = 2.f * p[dstOff + c] - p[refOff + c];
}

// layer-0 dense part: out = relu(TxAll0[n,18] @ W0[18,32] + b0)  (too skinny for WMMA)
__global__ void gemm0_relu_kernel(const float* __restrict__ tx, const float* __restrict__ W,
                                  const float* __restrict__ b, float* __restrict__ out, int n) {
  int i = blockIdx.x * blockDim.x + threadIdx.x;
  if (i >= n) return;
  float a[18];
#pragma unroll
  for (int j = 0; j < 18; ++j) a[j] = tx[i * 18 + j];
#pragma unroll
  for (int f = 0; f < 32; ++f) {
    float s = b[f];
#pragma unroll
    for (int j = 0; j < 18; ++j) s = fmaf(a[j], W[j * 32 + f], s);
    out[i * 32 + f] = fmaxf(s, 0.f);
  }
}

// pooling scatter: dst[:, 0..C) += vals[i] * x[cols[i], :]   (dst = plane 0 of next TxAll)
__global__ void pool_kernel(const int* __restrict__ rows, const int* __restrict__ cols,
                            const float* __restrict__ vals, const float* __restrict__ x,
                            float* __restrict__ dst, int nnz, int C, int dstStride) {
  int t = blockIdx.x * blockDim.x + threadIdx.x;
  if (t >= nnz * C) return;
  int i = t / C, c = t - i * C;
  atomicAdd(&dst[rows[i] * dstStride + c], vals[i] * x[cols[i] * C + c]);
}

// convert + transpose weights: W[KK,F] f32  ->  Wt[F,KK] f16
__global__ void wcvt_kernel(const float* __restrict__ W, _Float16* __restrict__ Wt,
                            int KK, int F) {
  int t = blockIdx.x * blockDim.x + threadIdx.x;
  if (t >= KK * F) return;
  int kk = t / F, f = t - kk * F;
  Wt[f * KK + kk] = (_Float16)W[kk * F + f];
}

// ---------------- WMMA GEMM: out[np,F] = (relu?)(A[np,KK] @ Wt^T + bias) ----------------
// A is f32 row-major [np, KK] with np a multiple of 16 (padded rows are zero).
// Wt is f16 [F, KK] (transposed weights). One wave per 16x16 output tile; no guards,
// EXEC is all-ones throughout (WMMA requirement). K-loop fully unrolled.
template <int KK, int F, int RELU>
__global__ void wmma_gemm_kernel(const float* __restrict__ A, const _Float16* __restrict__ Wt,
                                 const float* __restrict__ bias, float* __restrict__ out,
                                 int nTiles) {
  const int lane = threadIdx.x & 31;
  const int wave = threadIdx.x >> 5;
  const int tile = blockIdx.x * (blockDim.x >> 5) + wave;
  if (tile >= nTiles) return;                 // wave-uniform exit only
  constexpr int fT = F >> 4;
  const int m0 = (tile / fT) * 16;
  const int f0 = (tile % fT) * 16;
  const int mr = lane & 15;                   // A row within tile / out column within tile
  const int hi = lane >> 4;

  const float4*  arow = (const float4*)(A + (size_t)(m0 + mr) * KK);
  const _Float16* wrow = Wt + (size_t)(f0 + mr) * KK;

  v8f acc = {};
#pragma unroll
  for (int kb = 0; kb < KK; kb += 32) {
    // A fragment (16-bit A 16x32 layout): halfs 0..7 -> K=8*hi+j ; 8..15 -> K=16+8*hi+j
    const float4* ap = arow + (kb >> 2);
    float4 p0 = ap[2 * hi];
    float4 p1 = ap[2 * hi + 1];
    float4 p2 = ap[4 + 2 * hi];
    float4 p3 = ap[4 + 2 * hi + 1];
    v16h a;
    a[0] = (_Float16)p0.x;  a[1] = (_Float16)p0.y;
    a[2] = (_Float16)p0.z;  a[3] = (_Float16)p0.w;
    a[4] = (_Float16)p1.x;  a[5] = (_Float16)p1.y;
    a[6] = (_Float16)p1.z;  a[7] = (_Float16)p1.w;
    a[8]  = (_Float16)p2.x; a[9]  = (_Float16)p2.y;
    a[10] = (_Float16)p2.z; a[11] = (_Float16)p2.w;
    a[12] = (_Float16)p3.x; a[13] = (_Float16)p3.y;
    a[14] = (_Float16)p3.z; a[15] = (_Float16)p3.w;
    // B fragment (32x16): halfs 0..7 -> K=8*hi+j ; 8..15 -> K=16+8*hi+j, N = mr
    typedef __attribute__((ext_vector_type(8))) _Float16 v8h_;
    v8h_ b0 = *(const v8h_*)(wrow + kb + 8 * hi);
    v8h_ b1 = *(const v8h_*)(wrow + kb + 16 + 8 * hi);
    v16h b = __builtin_shufflevector(b0, b1, 0, 1, 2, 3, 4, 5, 6, 7, 8, 9, 10, 11, 12, 13, 14, 15);

    acc = __builtin_amdgcn_wmma_f32_16x16x32_f16(false, a, false, b, (short)0, acc,
                                                 false, false);
  }
  // C/D layout: VGPR r -> row (r + 8*hi), col = lane&15 ; padded out rows absorb overflow
  const float bi = bias[f0 + mr];
#pragma unroll
  for (int r = 0; r < 8; ++r) {
    int row = m0 + r + 8 * hi;
    float v = acc[r] + bi;
    if (RELU) v = fmaxf(v, 0.f);
    out[(size_t)row * F + f0 + mr] = v;
  }
}

// ---------------- final linear: Z = lin_w[10, M] @ x + lin_b ----------------
__global__ void final_init_kernel(const float* __restrict__ lb, float* __restrict__ out) {
  int i = threadIdx.x;
  if (i < kCls) out[i] = lb[i];
}

__global__ void final_reduce_kernel(const float* __restrict__ lw, const float* __restrict__ x,
                                    float* __restrict__ out, int M) {
  __shared__ float sm[256];
  float acc[kCls];
#pragma unroll
  for (int c = 0; c < kCls; ++c) acc[c] = 0.f;
  const int stride = gridDim.x * blockDim.x;
  for (int j = blockIdx.x * blockDim.x + threadIdx.x; j < M; j += stride) {
    float xv = x[j];
    __builtin_prefetch(&x[j + stride], 0, 1);   // global_prefetch hint
#pragma unroll
    for (int c = 0; c < kCls; ++c) acc[c] = fmaf(lw[c * M + j], xv, acc[c]);
  }
  for (int c = 0; c < kCls; ++c) {
    sm[threadIdx.x] = acc[c];
    __syncthreads();
    for (int s = 128; s > 0; s >>= 1) {
      if (threadIdx.x < (unsigned)s) sm[threadIdx.x] += sm[threadIdx.x + s];
      __syncthreads();
    }
    if (threadIdx.x == 0) atomicAdd(&out[c], sm[0]);
    __syncthreads();
  }
}

// ---------------- host-side helpers ----------------
static inline int cdiv(int a, int b) { return (a + b - 1) / b; }

static void run_cheb_props(const int* rows, const int* cols, int E, int n, int C, int stride,
                           float* dinv, float* tx, hipStream_t stream) {
  const int B = 256;
  zero_kernel<<<cdiv(n, B), B, 0, stream>>>(dinv, n);
  deg_kernel<<<cdiv(E, B), B, 0, stream>>>(rows, dinv, E);
  dinv_kernel<<<cdiv(n, B), B, 0, stream>>>(dinv, n);
  // Tx1 = prop(Tx0)
  prop_kernel<<<cdiv(E * C, B), B, 0, stream>>>(rows, cols, dinv, tx, E, C, stride, 0, C);
  for (int k = 2; k < kK; ++k) {
    prop_kernel<<<cdiv(E * C, B), B, 0, stream>>>(rows, cols, dinv, tx, E, C, stride,
                                                  (k - 1) * C, k * C);
    fixup_kernel<<<cdiv(n * C, B), B, 0, stream>>>(tx, n, C, stride, k * C, (k - 2) * C);
  }
}

extern "C" void kernel_launch(void* const* d_in, const int* in_sizes, int n_in,
                              void* d_out, int out_size, void* d_ws, size_t ws_size,
                              hipStream_t stream) {
  (void)in_sizes; (void)n_in; (void)out_size; (void)ws_size;
  const float* pos = (const float*)d_in[0];
  const int*   ei0 = (const int*)d_in[1];
  const int*   ei1 = (const int*)d_in[2];
  const int*   ei2 = (const int*)d_in[3];
  const int*   d0r = (const int*)d_in[4];
  const int*   d0c = (const int*)d_in[5];
  const float* d0v = (const float*)d_in[6];
  const int*   d1r = (const int*)d_in[7];
  const int*   d1c = (const int*)d_in[8];
  const float* d1v = (const float*)d_in[9];
  const float* W0  = (const float*)d_in[10];
  const float* b0  = (const float*)d_in[11];
  const float* W1  = (const float*)d_in[12];   // flat (6,32,64) == [192,64] row-major
  const float* b1  = (const float*)d_in[13];
  const float* W2  = (const float*)d_in[14];   // flat (6,64,128) == [384,128] row-major
  const float* b2  = (const float*)d_in[15];
  const float* lw  = (const float*)d_in[16];
  const float* lb  = (const float*)d_in[17];
  float* out = (float*)d_out;

  // workspace layout (floats), all chunks multiple-of-4 floats (16B aligned)
  float* ws   = (float*)d_ws;
  float* dinv = ws;                          // kN0
  float* tx0  = dinv + kN0;                  // kN0 * 18
  float* x0   = tx0  + kN0 * 18;             // kN0 * 32
  float* tx1  = x0   + kN0 * 32;             // kN1p * 192
  float* x1   = tx1  + kN1p * 192;           // kN1p * 64
  float* tx2  = x1   + kN1p * 64;            // kN2p * 384
  float* x2   = tx2  + kN2p * 384;           // kN2p * 128
  _Float16* Wt1 = (_Float16*)(x2 + kN2p * 128);   // 64*192 halfs
  _Float16* Wt2 = Wt1 + 64 * 192;                 // 128*384 halfs

  const int B = 256;

  // weight convert/transpose (tiny)
  wcvt_kernel<<<cdiv(192 * 64, B), B, 0, stream>>>(W1, Wt1, 192, 64);
  wcvt_kernel<<<cdiv(384 * 128, B), B, 0, stream>>>(W2, Wt2, 384, 128);

  // ---------- layer 0: ChebConv(3 -> 32) on N0 graph + ReLU ----------
  zero_kernel<<<cdiv(kN0 * 18, B), B, 0, stream>>>(tx0, kN0 * 18);
  seed0_kernel<<<cdiv(kN0 * 3, B), B, 0, stream>>>(pos, tx0, kN0, 18);
  run_cheb_props(ei0, ei0 + kE0, kE0, kN0, 3, 18, dinv, tx0, stream);
  gemm0_relu_kernel<<<cdiv(kN0, B), B, 0, stream>>>(tx0, W0, b0, x0, kN0);

  // ---------- pool 0: N0 -> N1, result lands in plane 0 of tx1 ----------
  zero_kernel<<<cdiv(kN1p * 192, B), B, 0, stream>>>(tx1, kN1p * 192);  // incl. padded rows
  pool_kernel<<<cdiv(kN0 * 32, B), B, 0, stream>>>(d0r, d0c, d0v, x0, tx1, kN0, 32, 192);

  // ---------- layer 1: ChebConv(32 -> 64) + ReLU, fused-K WMMA GEMM ----------
  run_cheb_props(ei1, ei1 + kE1, kE1, kN1, 32, 192, dinv, tx1, stream);
  {
    int tiles = (kN1p / 16) * (64 / 16);
    wmma_gemm_kernel<192, 64, 1><<<cdiv(tiles, 8), 256, 0, stream>>>(tx1, Wt1, b1, x1, tiles);
  }

  // ---------- pool 1: N1 -> N2 ----------
  zero_kernel<<<cdiv(kN2p * 384, B), B, 0, stream>>>(tx2, kN2p * 384);  // incl. padded rows
  pool_kernel<<<cdiv(kN1 * 64, B), B, 0, stream>>>(d1r, d1c, d1v, x1, tx2, kN1, 64, 384);

  // ---------- layer 2: ChebConv(64 -> 128), no ReLU ----------
  run_cheb_props(ei2, ei2 + kE2, kE2, kN2, 64, 384, dinv, tx2, stream);
  {
    int tiles = (kN2p / 16) * (128 / 16);
    wmma_gemm_kernel<384, 128, 0><<<cdiv(tiles, 8), 256, 0, stream>>>(tx2, Wt2, b2, x2, tiles);
  }

  // ---------- final linear: Z = lin_w @ x2.flatten() + lin_b ----------
  // x2 rows 0..kN2-1 are contiguous, so the flat view matches reshape(-1)
  final_init_kernel<<<1, 32, 0, stream>>>(lb, out);
  final_reduce_kernel<<<256, 256, 0, stream>>>(lw, x2, out, kN2 * 128);
}
